// TFData2VecVisionSelfAttention_16999480558394
// MI455X (gfx1250) — compile-verified, matrix operand order checked by reading, not run on hardware
//
#include <hip/hip_runtime.h>
#include <stdint.h>

// Problem constants
#define B_  8
#define S_  1025
#define D_  768
#define H_  12
#define HD_ 64
#define SP2_ 1152         // S padded to 9*128: every proj tile row is storable
#define NKB_ 17           // ceil(1040/64) key blocks (keys beyond S_ masked)
#define SCALE_ 0.125f     // 1/sqrt(64)

typedef __attribute__((ext_vector_type(16))) __bf16        v16bf;
typedef __attribute__((ext_vector_type(8)))  float         v8f;
typedef __attribute__((ext_vector_type(8)))  unsigned int  v8u;

// gcc-style vector types matching the builtin signatures
typedef int          v4i   __attribute__((vector_size(16)));
typedef unsigned int u32x4 __attribute__((vector_size(16)));
typedef int          i32x8 __attribute__((vector_size(32)));
typedef int          i32x4 __attribute__((vector_size(16)));

typedef __attribute__((address_space(1))) v4i as1_v4i;   // global int4*
typedef __attribute__((address_space(3))) v4i as3_v4i;   // LDS int4*

#if __has_builtin(__builtin_amdgcn_global_load_async_to_lds_b128) && \
    __has_builtin(__builtin_amdgcn_s_wait_asynccnt)
#define ASYNC_OK 1
#endif
#if __has_builtin(__builtin_amdgcn_tensor_load_to_lds) && \
    __has_builtin(__builtin_amdgcn_s_wait_tensorcnt)
#define TDM_OK 1
#endif

static __device__ __forceinline__ int imin(int a, int b) { return a < b ? a : b; }

// fp32 -> bf16 round-to-nearest-even
static __device__ __forceinline__ uint16_t f2bf(float f) {
  uint32_t u = __builtin_bit_cast(uint32_t, f);
  u += 0x7FFFu + ((u >> 16) & 1u);
  return (uint16_t)(u >> 16);
}
static __device__ __forceinline__ uint32_t pack_bf16(float f0, float f1) {
  return (uint32_t)f2bf(f0) | ((uint32_t)f2bf(f1) << 16);
}

// dword index (K-pair) within a 16-bit A/B fragment row, per CDNA5 ISA 7.12.2
static __device__ __forceinline__ int fragIdx(int g, int half) {
  return (g < 4 ? g : g + 4) + 4 * half;
}

// ---------------------------------------------------------------------------
// Flash-style attention per (b,h), 128 query rows per workgroup.
// Double-buffered K / V^T tiles: K staged by the Tensor Data Mover (one 2D
// descriptor), V^T staged by async global->LDS b128 DMA. Per iteration:
// wait counters -> barrier -> issue next tile's DMA -> WMMA on current tile.
// ---------------------------------------------------------------------------
__global__ __launch_bounds__(256)
void attn_kernel(const uint16_t* __restrict__ Q,   // [bh][SP2_][64] bf16
                 const uint16_t* __restrict__ K,   // [bh][SP2_][64] bf16
                 const uint16_t* __restrict__ Vt,  // [bh][64][SP2_] bf16
                 const float* __restrict__ bias,   // [1,H,S,S]
                 float* __restrict__ out)          // [B,S,D]
{
  __shared__ __attribute__((aligned(16))) uint32_t lds_k [2][64 * 32]; // 64 key x 64 hd bf16
  __shared__ __attribute__((aligned(16))) uint32_t lds_vt[2][64 * 32]; // 64 hd x 64 key bf16
  __shared__ __attribute__((aligned(16))) uint16_t lds_p[8][16 * 64];  // per-wave P tile

  const int qb = blockIdx.x;
  const int h  = blockIdx.y;
  const int b  = blockIdx.z;
  const int bh = b * H_ + h;

  const int t    = threadIdx.x;
  const int lane = t & 31;
  const int w    = t >> 5;
  const int half = lane >> 4;
  const int l16  = lane & 15;

  const uint32_t* Qu = (const uint32_t*)Q;
  const uint32_t* Ku = (const uint32_t*)K;
  const uint32_t* Vu = (const uint32_t*)Vt;

  // Issue one 64-key block's staging DMA into buffer `sel`
  auto issue = [&](int kbn, int sel) {
#ifdef TDM_OK
    // K tile via TDM: 2D descriptor, 64 rows x 128B, row stride 128B.
    if (w == 0) {
      uint64_t gaddr = (uint64_t)(uintptr_t)(Ku + ((size_t)bh * SP2_ + kbn) * 32);
      uint32_t laddr = (uint32_t)(uintptr_t)&lds_k[sel][0];
      u32x4 g0 = { 1u,                 // count=1, user descriptor
                   laddr,              // lds_addr
                   (uint32_t)gaddr,    // global_addr[31:0]
                   (uint32_t)((gaddr >> 32) & 0x01FFFFFFull) | (2u << 30) }; // [56:32]|type=2
      i32x8 g1 = { (int)(3u << 16),    // workgroup_mask=0, data_size=3 (8B units)
                   (int)(16u << 16),   // tensor_dim0[15:0]=16 units/row
                   (int)(64u << 16),   // tensor_dim1[15:0]=64 rows
                   (int)(16u << 16),   // tile_dim0=16
                   64,                 // tile_dim1=64, tile_dim2=0
                   16,                 // tensor_dim0_stride=16 units (128B)
                   0, 0 };
      i32x4 g2 = { 0, 0, 0, 0 };
      i32x4 g3 = { 0, 0, 0, 0 };
      i32x8 g4 = { 0, 0, 0, 0, 0, 0, 0, 0 };
      __builtin_amdgcn_tensor_load_to_lds(g0, g1, g2, g3, g4, 0);
    }
#elif defined(ASYNC_OK)
    #pragma unroll
    for (int i = 0; i < 2; ++i) {
      int c    = i * 256 + t;            // b128 chunk id 0..511
      int key  = c >> 3;
      int coff = (c & 7) * 4;
      const uint32_t* src = Ku + ((size_t)bh * SP2_ + kbn + key) * 32 + coff;
      __builtin_amdgcn_global_load_async_to_lds_b128(
          (as1_v4i*)(uintptr_t)src,
          (as3_v4i*)(uint32_t)(uintptr_t)&lds_k[sel][c * 4], 0, 0);
    }
#else
    #pragma unroll
    for (int i = 0; i < 8; ++i) {
      int idx = i * 256 + t;
      int key = idx >> 5, off = idx & 31;
      lds_k[sel][idx] = Ku[((size_t)bh * SP2_ + kbn + key) * 32 + off];
    }
#endif

#ifdef ASYNC_OK
    // V^T tile via async global->LDS DMA (ASYNCcnt-tracked)
    #pragma unroll
    for (int i = 0; i < 2; ++i) {
      int c    = i * 256 + t;
      int hd   = c >> 3;
      int coff = (c & 7) * 4;
      const uint32_t* src = Vu + ((size_t)bh * HD_ + hd) * (SP2_ / 2) + kbn / 2 + coff;
      __builtin_amdgcn_global_load_async_to_lds_b128(
          (as1_v4i*)(uintptr_t)src,
          (as3_v4i*)(uint32_t)(uintptr_t)&lds_vt[sel][c * 4], 0, 0);
    }
#else
    #pragma unroll
    for (int i = 0; i < 8; ++i) {
      int idx = i * 256 + t;
      int hd = idx >> 5, off = idx & 31;
      lds_vt[sel][idx] = Vu[((size_t)bh * HD_ + hd) * (SP2_ / 2) + kbn / 2 + off];
    }
#endif
  };

  // Q fragments for this wave's 16 rows (A layout: lane row = l16), 2 K-steps
  const int qrow = qb * 128 + 16 * w + l16;      // < SP2_, always in-bounds
  v16bf qa[2];
  #pragma unroll
  for (int ks = 0; ks < 2; ++ks) {
    v8u au;
    #pragma unroll
    for (int g = 0; g < 8; ++g)
      au[g] = Qu[((size_t)bh * SP2_ + qrow) * 32 + ks * 16 + fragIdx(g, half)];
    qa[ks] = __builtin_bit_cast(v16bf, au);
  }

  float mrow[8], lrow[8];
  int   qg[8];
  #pragma unroll
  for (int r = 0; r < 8; ++r) {
    mrow[r] = -3.0e38f;
    lrow[r] = 0.0f;
    qg[r]   = imin(qb * 128 + 16 * w + r + 8 * half, S_ - 1); // C-layout row (bias idx)
  }
  v8f ctx[4];
  #pragma unroll
  for (int ct = 0; ct < 4; ++ct)
    #pragma unroll
    for (int e = 0; e < 8; ++e) ctx[ct][e] = 0.0f;

  const float* biasH = bias + (size_t)h * S_ * S_;

  issue(0, 0);   // prologue: stage first block

  for (int it = 0; it < NKB_; ++it) {
    const int kb  = it * 64;
    const int sel = it & 1;

    // Wait for this iteration's staged tiles, then make them visible.
#ifdef ASYNC_OK
    __builtin_amdgcn_s_wait_asynccnt(0);
#endif
#ifdef TDM_OK
    if (w == 0) __builtin_amdgcn_s_wait_tensorcnt(0);
#endif
    __syncthreads();

    // Kick off next block's DMA into the other buffer (overlaps compute).
    if (it + 1 < NKB_) issue(kb + 64, sel ^ 1);

    // Scores: 16 x 64, 2 K-steps x 4 col tiles
    v8f sc[4];
    #pragma unroll
    for (int ct = 0; ct < 4; ++ct)
      #pragma unroll
      for (int e = 0; e < 8; ++e) sc[ct][e] = 0.0f;
    #pragma unroll
    for (int ks = 0; ks < 2; ++ks) {
      #pragma unroll
      for (int ct = 0; ct < 4; ++ct) {
        v8u bu;
        #pragma unroll
        for (int g = 0; g < 8; ++g)
          bu[g] = lds_k[sel][(16 * ct + l16) * 32 + ks * 16 + fragIdx(g, half)];
        v16bf kb16 = __builtin_bit_cast(v16bf, bu);
        sc[ct] = __builtin_amdgcn_wmma_f32_16x16x32_bf16(
            false, qa[ks], false, kb16, (short)0, sc[ct], false, false);
      }
    }

    // Online softmax per row (C layout: lane holds col l16, rows r+8*half)
    #pragma unroll
    for (int r = 0; r < 8; ++r) {
      float x[4];
      #pragma unroll
      for (int ct = 0; ct < 4; ++ct) {
        int key = kb + 16 * ct + l16;
        float bb = biasH[(size_t)qg[r] * S_ + imin(key, S_ - 1)];
        float v = sc[ct][r] * SCALE_ + bb;
        x[ct] = (key < S_) ? v : -3.0e38f;
      }
      float tm = fmaxf(fmaxf(x[0], x[1]), fmaxf(x[2], x[3]));
      #pragma unroll
      for (int msk = 1; msk < 16; msk <<= 1)
        tm = fmaxf(tm, __shfl_xor(tm, msk, 32));
      float mnew  = fmaxf(mrow[r], tm);
      float alpha = __expf(mrow[r] - mnew);
      float psum  = 0.0f;
      #pragma unroll
      for (int ct = 0; ct < 4; ++ct) {
        float p = __expf(x[ct] - mnew);
        psum += p;
        lds_p[w][(r + 8 * half) * 64 + 16 * ct + l16] = f2bf(p);
      }
      #pragma unroll
      for (int msk = 1; msk < 16; msk <<= 1)
        psum += __shfl_xor(psum, msk, 32);
      lrow[r] = lrow[r] * alpha + psum;
      mrow[r] = mnew;
      #pragma unroll
      for (int ct = 0; ct < 4; ++ct) ctx[ct][r] *= alpha;
    }

    // ctx += P @ V  (P re-read from this wave's LDS region in A layout)
    const uint32_t* pw = (const uint32_t*)&lds_p[w][0];
    #pragma unroll
    for (int ks = 0; ks < 2; ++ks) {
      v8u au;
      #pragma unroll
      for (int g = 0; g < 8; ++g)
        au[g] = pw[l16 * 32 + ks * 16 + fragIdx(g, half)];
      v16bf pa = __builtin_bit_cast(v16bf, au);
      #pragma unroll
      for (int ct = 0; ct < 4; ++ct) {
        v8u bu;
        #pragma unroll
        for (int g = 0; g < 8; ++g)
          bu[g] = lds_vt[sel][(16 * ct + l16) * 32 + ks * 16 + fragIdx(g, half)];
        v16bf vb = __builtin_bit_cast(v16bf, bu);
        ctx[ct] = __builtin_amdgcn_wmma_f32_16x16x32_bf16(
            false, pa, false, vb, (short)0, ctx[ct], false, false);
      }
    }
  }

  // Normalize and store fp32 output [B,S,D]
  #pragma unroll
  for (int ct = 0; ct < 4; ++ct) {
    int hd = 16 * ct + l16;
    #pragma unroll
    for (int r = 0; r < 8; ++r) {
      int s = qb * 128 + 16 * w + r + 8 * half;
      if (s < S_) {
        out[((size_t)b * S_ + s) * D_ + h * HD_ + hd] = ctx[ct][r] / lrow[r];
      }
    }
  }
}

// ---------------------------------------------------------------------------
// QKV projection: X[B,S,D] @ W[D,D] + bias -> bf16, head-major layouts.
//   TR==0 : out[bh][s_pad][hd]   (q, k)
//   TR==1 : out[bh][hd][s_pad]   (v, pre-transposed for PV B-fragments)
// SP2_ padding makes every tile row storable -> unguarded epilogue stores.
// ---------------------------------------------------------------------------
template <int TR>
__global__ __launch_bounds__(256)
void qkv_proj_kernel(const float* __restrict__ X,
                     const float* __restrict__ W,
                     const float* __restrict__ bias,
                     uint16_t* __restrict__ out)
{
  __shared__ __attribute__((aligned(16))) uint32_t lds_a[128 * 16]; // 128 x 32 bf16
  __shared__ __attribute__((aligned(16))) uint32_t lds_b[64 * 16];  // 64(n) x 32(k) bf16

  const int mb = blockIdx.x;
  const int nb = blockIdx.y;     // head index
  const int b  = blockIdx.z;

  const int t    = threadIdx.x;
  const int lane = t & 31;
  const int w    = t >> 5;       // wave 0..7, owns rows 16w..16w+15 of the tile
  const int half = lane >> 4;
  const int l16  = lane & 15;

  v8f acc[4];
  #pragma unroll
  for (int ct = 0; ct < 4; ++ct)
    #pragma unroll
    for (int e = 0; e < 8; ++e) acc[ct][e] = 0.0f;

  for (int k0 = 0; k0 < D_; k0 += 32) {
    // Stage A tile (branch-free clamp+select)
    #pragma unroll
    for (int i = 0; i < 8; ++i) {
      int idx = i * 256 + t;             // dword index, 2048 total
      int row = idx >> 4;
      int cc  = (idx & 15) * 2;
      int s   = mb * 128 + row;
      int sc  = imin(s, S_ - 1);
      float2 f = *(const float2*)(X + ((size_t)b * S_ + sc) * D_ + k0 + cc);
      if (s >= S_) { f.x = 0.0f; f.y = 0.0f; }    // cndmask, no exec branch
      lds_a[idx] = pack_bf16(f.x, f.y);
    }
    // Stage B tile transposed: coalesced float2 row-major loads
    uint16_t* lb16 = (uint16_t*)lds_b;
    #pragma unroll
    for (int i = 0; i < 4; ++i) {
      int idx = i * 256 + t;             // 1024 float2: 32 k-rows x 32 n-pairs
      int k   = idx >> 5;
      int n0  = (idx & 31) * 2;
      float2 f = *(const float2*)(W + (size_t)(k0 + k) * D_ + nb * 64 + n0);
      lb16[(n0    ) * 32 + k] = f2bf(f.x);
      lb16[(n0 + 1) * 32 + k] = f2bf(f.y);
    }
    __syncthreads();

    v8u au;
    #pragma unroll
    for (int g = 0; g < 8; ++g)
      au[g] = lds_a[(16 * w + l16) * 16 + fragIdx(g, half)];
    v16bf a = __builtin_bit_cast(v16bf, au);

    #pragma unroll
    for (int ct = 0; ct < 4; ++ct) {
      v8u bu;
      #pragma unroll
      for (int g = 0; g < 8; ++g)
        bu[g] = lds_b[(16 * ct + l16) * 16 + fragIdx(g, half)];
      v16bf bmat = __builtin_bit_cast(v16bf, bu);
      acc[ct] = __builtin_amdgcn_wmma_f32_16x16x32_bf16(
          false, a, false, bmat, (short)0, acc[ct], false, false);
    }
    __syncthreads();
  }

  const int bh = b * H_ + nb;
  #pragma unroll
  for (int ct = 0; ct < 4; ++ct) {
    int hd = 16 * ct + l16;
    float bv = bias[nb * 64 + hd];
    #pragma unroll
    for (int r = 0; r < 8; ++r) {
      int s = mb * 128 + 16 * w + r + 8 * half;   // < SP2_ always
      uint16_t val = f2bf(acc[ct][r] + bv);
      if (TR == 0)
        out[((size_t)bh * SP2_ + s) * HD_ + hd] = val;
      else
        out[((size_t)bh * HD_ + hd) * SP2_ + s] = val;
    }
  }
}

// ---------------------------------------------------------------------------
extern "C" void kernel_launch(void* const* d_in, const int* in_sizes, int n_in,
                              void* d_out, int out_size, void* d_ws, size_t ws_size,
                              hipStream_t stream) {
  const float* hs = (const float*)d_in[0];
  const float* rb = (const float*)d_in[1];
  const float* Wq = (const float*)d_in[2];
  const float* bq = (const float*)d_in[3];
  const float* Wk = (const float*)d_in[4];
  const float* bk = (const float*)d_in[5];
  const float* Wv = (const float*)d_in[6];
  const float* bv = (const float*)d_in[7];
  float* out = (float*)d_out;

  const size_t perBuf = (size_t)B_ * H_ * SP2_ * HD_;  // bf16 elements per buffer
  uint16_t* qws = (uint16_t*)d_ws;
  uint16_t* kws = qws + perBuf;
  uint16_t* vws = kws + perBuf;

  dim3 gp(9, H_, B_), bp(256);
  qkv_proj_kernel<0><<<gp, bp, 0, stream>>>(hs, Wq, bq, qws);
  qkv_proj_kernel<0><<<gp, bp, 0, stream>>>(hs, Wk, bk, kws);
  qkv_proj_kernel<1><<<gp, bp, 0, stream>>>(hs, Wv, bv, vws);

  dim3 ga(9, H_, B_), ba(256);
  attn_kernel<<<ga, ba, 0, stream>>>(qws, kws, vws, rb, out);
}
